// Logos44Field_51573967290979
// MI455X (gfx1250) — matrix operand order repaired and checked by reading, not compiled, not run on hardware
//
#include <hip/hip_runtime.h>

// ---------------------------------------------------------------------------
// Fused 44-layer "Logos field" network for MI455X (gfx1250, wave32, WMMA).
// All matmuls via v_wmma_f32_16x16x32_f16. Each wave owns a 16-token tile and
// runs all 44 layers with h resident in LDS (no HBM round trips for h).
// Weights are repacked once into fragment-ordered f16 B-panels in d_ws.
// ---------------------------------------------------------------------------

typedef _Float16 f16x16 __attribute__((ext_vector_type(16)));
typedef _Float16 f16x8  __attribute__((ext_vector_type(8)));
typedef float    f32x8  __attribute__((ext_vector_type(8)));
typedef float    f32x4  __attribute__((ext_vector_type(4)));

#define L_DEPTH 44
#define L_SCALE 4.0f
#define DIM   256
#define RANK  32
#define NSIG  128
#define VOCAB 512
#define TLEN  512
#define NTOK  (128 * 512)

// ---- d_ws layout (in _Float16 units). B-panel tile = 32(K) x 16(N) = 512 halves.
// fragment order inside a tile:  idx = lane*16 + j,
//   K = 32*kt + (lane>>4)*8 + (j&7) + (j>>3)*16 ,  N = 16*nt + (lane&15),  tile = kt*Ntiles + nt
#define OFF_PROJ  0u                      // K=256 N=64  -> 8*4*512   = 16384
#define OFF_K2C   16384u                  // K=32  N=32  -> 1*2*512   = 1024
#define OFF_CODES 17408u                  // K=32  N=128 -> 1*8*512   = 4096
#define OFF_SIG   21504u                  // K=128 N=256 -> 4*16*512  = 32768
#define OFF_UP    54272u                  // K=32  N=256 -> 1*16*512  = 8192
#define OFF_GATE  62464u                  // K=512 N=256 -> 16*16*512 = 131072
#define OFF_EMB   193536u                 // K=256 N=512 -> 8*32*512  = 131072
// total 324608 halves ~= 634 KB of workspace

// ---------------------------------------------------------------------------
// Prep: repack a weight matrix into fragment-ordered f16 B panels.
//   B[k][n] = transpose ? src[n*srcCols + k] : src[k*srcCols + n]
// ---------------------------------------------------------------------------
__global__ void pack_b_panels(const float* __restrict__ src, _Float16* __restrict__ dst,
                              int K, int N, int srcCols, int transpose) {
  const int Ntiles = N >> 4;
  const int total  = (K >> 5) * Ntiles * 512;
  for (int idx = blockIdx.x * blockDim.x + threadIdx.x; idx < total;
       idx += gridDim.x * blockDim.x) {
    int tile = idx >> 9;
    int w    = idx & 511;
    int lane = w >> 4;
    int j    = w & 15;
    int kt   = tile / Ntiles;
    int nt   = tile - kt * Ntiles;
    int k    = (kt << 5) + ((lane >> 4) << 3) + (j & 7) + ((j >> 3) << 4);
    int n    = (nt << 4) + (lane & 15);
    float v  = transpose ? src[(size_t)n * srcCols + k] : src[(size_t)k * srcCols + n];
    dst[idx] = (_Float16)v;
  }
}

// ---------------------------------------------------------------------------
// Fragment helpers (layouts per CDNA5 ISA §7.12.2, wave32)
// ---------------------------------------------------------------------------
__device__ __forceinline__ f32x8 wmma_f16(f16x16 a, f16x16 b, f32x8 c) {
  return __builtin_amdgcn_wmma_f32_16x16x32_f16(false, a, false, b, (short)0, c,
                                                false, false);
}

// A fragment (16x32 f16) from a row-major f16 LDS buffer [16][stride]
__device__ __forceinline__ f16x16 loadA16(const _Float16* buf, int stride, int m,
                                          int hi, int k0) {
  const _Float16* p = buf + m * stride + k0 + hi * 8;
  f16x8 lo = *(const f16x8*)p;          // K = k0 + hi*8 .. +7
  f16x8 hh = *(const f16x8*)(p + 16);   // K = k0 + 16 + hi*8 .. +7
  f16x16 a;
#pragma unroll
  for (int i = 0; i < 8; ++i) { a[i] = lo[i]; a[i + 8] = hh[i]; }
  return a;
}

// A fragment from a row-major f32 LDS buffer, converted to f16 on the fly
__device__ __forceinline__ f16x16 loadA32(const float* buf, int stride, int m,
                                          int hi, int k0) {
  const float* p = buf + m * stride + k0 + hi * 8;
  f32x4 x0 = *(const f32x4*)p;
  f32x4 x1 = *(const f32x4*)(p + 4);
  f32x4 x2 = *(const f32x4*)(p + 16);
  f32x4 x3 = *(const f32x4*)(p + 20);
  f16x16 a;
#pragma unroll
  for (int i = 0; i < 4; ++i) {
    a[i]      = (_Float16)x0[i];
    a[i + 4]  = (_Float16)x1[i];
    a[i + 8]  = (_Float16)x2[i];
    a[i + 12] = (_Float16)x3[i];
  }
  return a;
}

// B fragment (32x16 f16) from a fragment-ordered global panel (L2-resident)
__device__ __forceinline__ f16x16 loadB(const _Float16* __restrict__ p, int tile,
                                        int lane) {
  return *(const f16x16*)(p + ((size_t)tile << 9) + (lane << 4));
}

// ---------------------------------------------------------------------------
// Main fused kernel: 128 threads (4 waves), each wave = 16 tokens, all layers.
// Dynamic LDS: 4KB norm params + 4 * 30KB per-wave scratch = 124 KB.
// ---------------------------------------------------------------------------
__global__ __launch_bounds__(128) void logos44_fused(
    const int* __restrict__ xidx, const float* __restrict__ embed,
    const float* __restrict__ phase, const float* __restrict__ normg,
    const float* __restrict__ normb, const float* __restrict__ angular,
    const float* __restrict__ gate_b, const float* __restrict__ outg,
    const float* __restrict__ outb, const _Float16* __restrict__ ws,
    float* __restrict__ out) {
  extern __shared__ __align__(16) char smem[];
  float* sNg = (float*)smem;        // 256 f32
  float* sNb = sNg + 256;
  float* sOg = sNb + 256;
  float* sOb = sOg + 256;           // norm area ends at 4096 B

  const _Float16* projP  = ws + OFF_PROJ;
  const _Float16* k2cP   = ws + OFF_K2C;
  const _Float16* codesP = ws + OFF_CODES;
  const _Float16* sigP   = ws + OFF_SIG;
  const _Float16* upP    = ws + OFF_UP;
  const _Float16* gateP  = ws + OFF_GATE;
  const _Float16* embP   = ws + OFF_EMB;

  const int wave = threadIdx.x >> 5;
  const int lane = threadIdx.x & 31;
  const int m    = lane & 15;   // row (M) / column-in-tile index
  const int hi   = lane >> 4;   // half-wave select

  char* wbase = smem + 4096 + (size_t)wave * 30720;
  float*    hW    = (float*)wbase;                // [16][256] f32  (16 KB)
  _Float16* mix16 = (_Float16*)(wbase + 16384);   // [16][256] f16  (8 KB) hn/mixed/hln
  _Float16* keyW  = (_Float16*)(wbase + 24576);   // [16][32]  f16
  _Float16* s2W   = (_Float16*)(wbase + 25600);   // [16][32]  f16
  _Float16* codeW = (_Float16*)(wbase + 26624);   // [16][128] f16  (4 KB)

  // Stage norm params once (block-shared)
  for (int i = threadIdx.x; i < 256; i += 128) {
    sNg[i] = normg[i]; sNb[i] = normb[i]; sOg[i] = outg[i]; sOb[i] = outb[i];
  }
  __syncthreads();

  const int tok0 = (blockIdx.x * 4 + wave) * 16;

  // ---- h0 = embed[x] + phase_enc[t]  (each lane fills half a token row)
  {
    const int tok = tok0 + m;
    const int xe  = xidx[tok];
    const int pos = tok & (TLEN - 1);
    const float* es = embed + (size_t)xe * DIM + hi * 128;
    const float* ps = phase + (size_t)pos * DIM + hi * 128;
    float* hd = hW + m * DIM + hi * 128;
#pragma unroll 1
    for (int i = 0; i < 128; i += 4) {
      f32x4 e = *(const f32x4*)(es + i);
      f32x4 p = *(const f32x4*)(ps + i);
      f32x4 r = e + p;
      *(f32x4*)(hd + i) = r;
    }
  }

  // per-lane constants hoisted out of the layer loop
  const float a0 = angular[m];
  const float a1 = angular[16 + m];
  float gb[16];
#pragma unroll
  for (int nt = 0; nt < 16; ++nt) gb[nt] = gate_b[nt * 16 + m];

  const f32x8 vzero = {0.f, 0.f, 0.f, 0.f, 0.f, 0.f, 0.f, 0.f};

#pragma unroll 1
  for (int layer = 0; layer < L_DEPTH; ++layer) {
    // ---- layer_norm(h) -> mix16 (hn, f16). Lane pair (m, m|hi) covers a row.
    {
      const float* p = hW + m * DIM + hi * 128;
      float s = 0.f, q = 0.f;
#pragma unroll 1
      for (int i = 0; i < 128; i += 4) {
        f32x4 v4 = *(const f32x4*)(p + i);
#pragma unroll
        for (int c = 0; c < 4; ++c) { s += v4[c]; q += v4[c] * v4[c]; }
      }
      s += __shfl_xor(s, 16, 32);
      q += __shfl_xor(q, 16, 32);
      const float mu  = s * (1.f / 256.f);
      const float var = q * (1.f / 256.f) - mu * mu;
      const float rs  = __frsqrt_rn(var + 1e-5f);
#pragma unroll 1
      for (int i = 0; i < 128; ++i) {
        const int d = hi * 128 + i;
        mix16[m * DIM + d] = (_Float16)((p[i] - mu) * rs * sNg[d] + sNb[d]);
      }
    }

    // ---- proj = hn @ proj_W^T : [16x256] x [256x64]
    f32x8 pacc[4];
#pragma unroll
    for (int n = 0; n < 4; ++n) pacc[n] = vzero;
#pragma unroll 1
    for (int k = 0; k < 8; ++k) {
      f16x16 a = loadA16(mix16, DIM, m, hi, k * 32);
#pragma unroll
      for (int n = 0; n < 4; ++n)
        pacc[n] = wmma_f16(a, loadB(projP, k * 4 + n, lane), pacc[n]);
    }

    // ---- key = sin(projA*ang) * cos(projB*ang) -> keyW [16x32] f16
#pragma unroll
    for (int j = 0; j < 2; ++j) {
      const float aj = j ? a1 : a0;
#pragma unroll
      for (int v = 0; v < 8; ++v) {
        const float kv = __sinf(pacc[j][v] * aj) * __cosf(pacc[j + 2][v] * aj);
        keyW[(v + hi * 8) * RANK + j * 16 + m] = (_Float16)kv;
      }
    }

    // ---- s2 = key @ k2c_W^T : [16x32] x [32x32]
    f32x8 kacc[2] = {vzero, vzero};
    {
      f16x16 a = loadA16(keyW, RANK, m, hi, 0);
#pragma unroll
      for (int n = 0; n < 2; ++n)
        kacc[n] = wmma_f16(a, loadB(k2cP, n, lane), kacc[n]);
    }
#pragma unroll
    for (int n = 0; n < 2; ++n)
#pragma unroll
      for (int v = 0; v < 8; ++v)
        s2W[(v + hi * 8) * RANK + n * 16 + m] = (_Float16)kacc[n][v];

    // ---- logits = s2 @ codes^T : [16x32] x [32x128]
    f32x8 lacc[8];
#pragma unroll
    for (int n = 0; n < 8; ++n) lacc[n] = vzero;
    {
      f16x16 a = loadA16(s2W, RANK, m, hi, 0);
#pragma unroll
      for (int n = 0; n < 8; ++n)
        lacc[n] = wmma_f16(a, loadB(codesP, n, lane), lacc[n]);
    }

    // ---- softmax over n_sig=128 (row spans 16 lanes x 8 acc tiles) -> codeW
#pragma unroll
    for (int v = 0; v < 8; ++v) {
      float mx = -3.0e38f;
#pragma unroll
      for (int j = 0; j < 8; ++j) {
        const float t = lacc[j][v] * L_SCALE;
        lacc[j][v] = t;
        mx = fmaxf(mx, t);
      }
      mx = fmaxf(mx, __shfl_xor(mx, 1, 32));
      mx = fmaxf(mx, __shfl_xor(mx, 2, 32));
      mx = fmaxf(mx, __shfl_xor(mx, 4, 32));
      mx = fmaxf(mx, __shfl_xor(mx, 8, 32));
      float ss = 0.f;
#pragma unroll
      for (int j = 0; j < 8; ++j) {
        const float e = __expf(lacc[j][v] - mx);
        lacc[j][v] = e;
        ss += e;
      }
      ss += __shfl_xor(ss, 1, 32);
      ss += __shfl_xor(ss, 2, 32);
      ss += __shfl_xor(ss, 4, 32);
      ss += __shfl_xor(ss, 8, 32);
      const float inv = 1.f / ss;
#pragma unroll
      for (int j = 0; j < 8; ++j)
        codeW[(v + hi * 8) * NSIG + j * 16 + m] = (_Float16)(lacc[j][v] * inv);
    }

    // ---- mixed = code @ signals + key @ up_W^T : -> macc [16x256]
    f32x8 macc[16];
#pragma unroll
    for (int n = 0; n < 16; ++n) macc[n] = vzero;
#pragma unroll 1
    for (int k = 0; k < 4; ++k) {
      f16x16 a = loadA16(codeW, NSIG, m, hi, k * 32);
#pragma unroll
      for (int n = 0; n < 16; ++n)
        macc[n] = wmma_f16(a, loadB(sigP, k * 16 + n, lane), macc[n]);
    }
    {
      f16x16 a = loadA16(keyW, RANK, m, hi, 0);
#pragma unroll
      for (int n = 0; n < 16; ++n)
        macc[n] = wmma_f16(a, loadB(upP, n, lane), macc[n]);
    }
    // mixed -> mix16 (hn is dead; reuse buffer)
#pragma unroll
    for (int n = 0; n < 16; ++n)
#pragma unroll
      for (int v = 0; v < 8; ++v)
        mix16[(v + hi * 8) * DIM + n * 16 + m] = (_Float16)macc[n][v];

    // ---- h_new = sigmoid([mixed | h] @ gate_W^T + gate_b) : K=512
    f32x8 gacc[16];
#pragma unroll
    for (int n = 0; n < 16; ++n) gacc[n] = vzero;
#pragma unroll 1
    for (int k = 0; k < 16; ++k) {
      f16x16 a = (k < 8) ? loadA16(mix16, DIM, m, hi, k * 32)
                         : loadA32(hW, DIM, m, hi, (k - 8) * 32);
#pragma unroll
      for (int n = 0; n < 16; ++n)
        gacc[n] = wmma_f16(a, loadB(gateP, k * 16 + n, lane), gacc[n]);
    }
    // sigmoid + bias, write back into LDS h (same wave: DS ops are in-order)
#pragma unroll
    for (int n = 0; n < 16; ++n)
#pragma unroll
      for (int v = 0; v < 8; ++v) {
        const float val = gacc[n][v] + gb[n];
        hW[(v + hi * 8) * DIM + n * 16 + m] = 1.f / (1.f + __expf(-val));
      }
  }

  // ---- final layer_norm(h, outn) -> mix16 (hln)
  {
    const float* p = hW + m * DIM + hi * 128;
    float s = 0.f, q = 0.f;
#pragma unroll 1
    for (int i = 0; i < 128; i += 4) {
      f32x4 v4 = *(const f32x4*)(p + i);
#pragma unroll
      for (int c = 0; c < 4; ++c) { s += v4[c]; q += v4[c] * v4[c]; }
    }
    s += __shfl_xor(s, 16, 32);
    q += __shfl_xor(q, 16, 32);
    const float mu  = s * (1.f / 256.f);
    const float var = q * (1.f / 256.f) - mu * mu;
    const float rs  = __frsqrt_rn(var + 1e-5f);
#pragma unroll 1
    for (int i = 0; i < 128; ++i) {
      const int d = hi * 128 + i;
      mix16[m * DIM + d] = (_Float16)((p[i] - mu) * rs * sOg[d] + sOb[d]);
    }
  }

  // ---- out = hln @ embed^T : [16x256] x [256x512] -> global f32
#pragma unroll 1
  for (int nt = 0; nt < 32; ++nt) {
    f32x8 oacc = vzero;
#pragma unroll 1
    for (int k = 0; k < 8; ++k) {
      f16x16 a = loadA16(mix16, DIM, m, hi, k * 32);
      oacc = wmma_f16(a, loadB(embP, k * 32 + nt, lane), oacc);
    }
#pragma unroll
    for (int v = 0; v < 8; ++v)
      out[(size_t)(tok0 + v + hi * 8) * VOCAB + nt * 16 + m] = oacc[v];
  }
}

// ---------------------------------------------------------------------------
extern "C" void kernel_launch(void* const* d_in, const int* in_sizes, int n_in,
                              void* d_out, int out_size, void* d_ws, size_t ws_size,
                              hipStream_t stream) {
  (void)in_sizes; (void)n_in; (void)out_size; (void)ws_size;
  const int*   x       = (const int*)d_in[0];
  const float* embed   = (const float*)d_in[1];   // [512][256]
  const float* phase   = (const float*)d_in[2];   // [512][256]
  const float* normg   = (const float*)d_in[3];
  const float* normb   = (const float*)d_in[4];
  const float* projW   = (const float*)d_in[5];   // [64][256]
  const float* angular = (const float*)d_in[6];   // [32]
  const float* upW     = (const float*)d_in[7];   // [256][32]
  const float* signals = (const float*)d_in[8];   // [128][256]
  const float* codes   = (const float*)d_in[9];   // [128][32]
  const float* k2cW    = (const float*)d_in[10];  // [32][32]
  const float* gateW   = (const float*)d_in[11];  // [256][512]
  const float* gateb   = (const float*)d_in[12];
  const float* outg    = (const float*)d_in[13];
  const float* outb    = (const float*)d_in[14];

  _Float16* ws = (_Float16*)d_ws;   // needs ~650 KB

  // Repack all weights into fragment-ordered f16 B-panels (L2-resident).
  dim3 pg(512), pb(256);
  pack_b_panels<<<pg, pb, 0, stream>>>(projW,   ws + OFF_PROJ,  256,  64, 256, 1);
  pack_b_panels<<<pg, pb, 0, stream>>>(k2cW,    ws + OFF_K2C,    32,  32,  32, 1);
  pack_b_panels<<<pg, pb, 0, stream>>>(codes,   ws + OFF_CODES,  32, 128,  32, 1);
  pack_b_panels<<<pg, pb, 0, stream>>>(signals, ws + OFF_SIG,   128, 256, 256, 0);
  pack_b_panels<<<pg, pb, 0, stream>>>(upW,     ws + OFF_UP,     32, 256,  32, 1);
  pack_b_panels<<<pg, pb, 0, stream>>>(gateW,   ws + OFF_GATE,  512, 256, 512, 1);
  pack_b_panels<<<pg, pb, 0, stream>>>(embed,   ws + OFF_EMB,   256, 512, 256, 1);

  // Fused network: 1024 blocks x 128 threads (4 waves x 16 tokens), 124 KB LDS.
  const size_t smem = 4096 + 4 * 30720;
  logos44_fused<<<dim3(NTOK / 64), dim3(128), smem, stream>>>(
      x, embed, phase, normg, normb, angular, gateb, outg, outb, ws,
      (float*)d_out);
}
